// GAT_34952443855370
// MI455X (gfx1250) — compile-verified, matrix operand order checked by reading, not run on hardware
//
#include <hip/hip_runtime.h>
#include <hip/hip_bf16.h>

typedef __attribute__((ext_vector_type(16))) _Float16 v16h;
typedef __attribute__((ext_vector_type(8)))  _Float16 v8h;
typedef __attribute__((ext_vector_type(4)))  _Float16 v4h;
typedef __attribute__((ext_vector_type(8)))  float    v8f;

#define NN    20000
#define NDEG  16
#define NHD   8
#define NC    128
#define NFIN  64
#define NOUT  16
#define HC    1024           // NHD * NC
#define SLOPE 0.01f
#define AGG_GRID  200
#define AGG_ITERS (NN / AGG_GRID)   // 100, exact

// ---------------------------------------------------------------- converts
__global__ void cvt_f32_f16(const float* __restrict__ in, _Float16* __restrict__ out, int n) {
    int i = blockIdx.x * blockDim.x + threadIdx.x;
    if (i < n) out[i] = (_Float16)in[i];
}

// ---------------------------------------------------------------- WMMA GEMM
// A fragment (16x32 f16, row-major source, row stride ld halves):
//   lane<16: row=lane,    holds K = k0 + {0..7, 16..23}
//   lane>=16: row=lane-16, holds K = k0 + {8..15, 24..31}
// B fragment uses the mirrored layout with "row" = output column (Bt is
// stored row-major as [out_col][k], i.e. pre-transposed weights).
__device__ __forceinline__ v16h load_frag(const _Float16* __restrict__ p, int ld, int k0, int lane) {
    int row = lane & 15;
    int kb  = k0 + ((lane >> 4) << 3);
    const _Float16* q = p + row * ld + kb;
    v8h lo = *(const v8h*)(q);        // K = kb .. kb+7
    v8h hi = *(const v8h*)(q + 16);   // K = kb+16 .. kb+23
    return __builtin_shufflevector(lo, hi, 0,1,2,3,4,5,6,7,8,9,10,11,12,13,14,15);
}

// H[n][hc] = sum_k A[n][k] * Bt[hc][k]; grid = (NN/16, 16), block = 128 (4 waves)
template <int K>
__global__ void proj_kernel(const _Float16* __restrict__ A,
                            const _Float16* __restrict__ Bt,
                            _Float16* __restrict__ Hm) {
    int lane  = threadIdx.x & 31;
    int wave  = threadIdx.x >> 5;
    int ntile = blockIdx.x;
    int ctile = blockIdx.y * 4 + wave;          // 0..63
    const _Float16* Ap = A  + (size_t)ntile * 16 * K;
    const _Float16* Bp = Bt + (size_t)ctile * 16 * K;
    v8f acc = {};
#pragma unroll
    for (int k0 = 0; k0 < K; k0 += 32) {
        v16h a = load_frag(Ap, K, k0, lane);
        v16h b = load_frag(Bp, K, k0, lane);
        acc = __builtin_amdgcn_wmma_f32_16x16x32_f16(false, a, false, b,
                                                     (short)0, acc, false, false);
    }
    // D layout: VGPR j -> row j (+8 for upper half-wave), col = lane&15
    int col   = ctile * 16 + (lane & 15);
    int rbase = ntile * 16 + ((lane >> 4) << 3);
#pragma unroll
    for (int j = 0; j < 8; ++j)
        Hm[(size_t)(rbase + j) * HC + col] = (_Float16)acc[j];
}

// ---------------------------------------------------------------- attention scores
// s_dst[n][h] = sum_c H[n][h*C+c] * a[h][c];  s_src uses a[h][C+c]
// block = 256 (8 waves, wave = head), grid = NN
__global__ void score_kernel(const _Float16* __restrict__ Hm, const float* __restrict__ a,
                             float* __restrict__ sdst, float* __restrict__ ssrc) {
    int n    = blockIdx.x;
    int h    = threadIdx.x >> 5;
    int lane = threadIdx.x & 31;
    const _Float16* hp = Hm + (size_t)n * HC + h * NC;
    const float* ad = a + h * (2 * NC);
    const float* as = ad + NC;
    float d_ = 0.f, s_ = 0.f;
    for (int c = lane; c < NC; c += 32) {
        float hv = (float)hp[c];
        d_ = fmaf(hv, ad[c], d_);
        s_ = fmaf(hv, as[c], s_);
    }
    for (int off = 16; off; off >>= 1) {
        d_ += __shfl_xor(d_, off, 32);
        s_ += __shfl_xor(s_, off, 32);
    }
    if (lane == 0) { sdst[n * NHD + h] = d_; ssrc[n * NHD + h] = s_; }
}

// ---------------------------------------------------------------- softmax + aggregate
// One 128-thread block per node.
// Phase 1 (logits): thread -> (h,d) = (tid>>4, tid&15), width-16 softmax groups.
// Phase 2 (gather): wave w owns heads {w, w+4}; thread owns 4 consecutive
//   channels (one b64 load per (head, neighbour)), then a 4-way LDS reduction
//   completes the 8-head mean per channel.
__device__ __forceinline__ float gat_aggregate(const _Float16* __restrict__ Hm,
                                               const float* __restrict__ sdst,
                                               const float* __restrict__ ssrc,
                                               const int* __restrict__ nbrs,
                                               int n, int tid,
                                               int* nb, float (*alpha)[NDEG],
                                               float (*red)[NC]) {
    if (tid < NDEG) nb[tid] = nbrs[n * NDEG + tid];
    __syncthreads();
    int h = tid >> 4, d = tid & 15;
    float e = sdst[n * NHD + h] + ssrc[nb[d] * NHD + h];
    e = (e > 0.f) ? e : SLOPE * e;                 // leaky_relu
    float m = e;
    for (int off = 8; off; off >>= 1) m = fmaxf(m, __shfl_xor(m, off, 16));
    float ex = __expf(e - m);
    float den = ex;
    for (int off = 8; off; off >>= 1) den += __shfl_xor(den, off, 16);
    alpha[h][d] = ex / den;
    __syncthreads();

    int hgrp  = tid >> 5;          // wave id: heads hgrp and hgrp+4
    int cbase = (tid & 31) * 4;    // 4 consecutive channels per thread
    float a0 = 0.f, a1 = 0.f, a2 = 0.f, a3 = 0.f;
#pragma unroll
    for (int hh = hgrp; hh < NHD; hh += 4) {
#pragma unroll
        for (int dd = 0; dd < NDEG; ++dd) {
            v4h hv = *(const v4h*)(Hm + (size_t)nb[dd] * HC + hh * NC + cbase);
            float al = alpha[hh][dd];
            a0 = fmaf(al, (float)hv[0], a0);
            a1 = fmaf(al, (float)hv[1], a1);
            a2 = fmaf(al, (float)hv[2], a2);
            a3 = fmaf(al, (float)hv[3], a3);
        }
    }
    red[hgrp][cbase + 0] = a0;
    red[hgrp][cbase + 1] = a1;
    red[hgrp][cbase + 2] = a2;
    red[hgrp][cbase + 3] = a3;
    __syncthreads();
    float total = (red[0][tid] + red[1][tid]) + (red[2][tid] + red[3][tid]);
    total *= (1.0f / NHD);
    float r = (total > 0.f) ? total : (__expf(total) - 1.0f);   // elu
    __syncthreads();   // protect nb/alpha/red for next iteration
    return r;
}

__global__ void attn1_kernel(const _Float16* __restrict__ Hm,
                             const float* __restrict__ sdst, const float* __restrict__ ssrc,
                             const int* __restrict__ nbrs, _Float16* __restrict__ embh) {
    __shared__ int   nb[NDEG];
    __shared__ float alpha[NHD][NDEG];
    __shared__ float red[4][NC];
    float r = gat_aggregate(Hm, sdst, ssrc, nbrs, blockIdx.x, threadIdx.x, nb, alpha, red);
    embh[(size_t)blockIdx.x * NC + threadIdx.x] = (_Float16)r;
}

// Layer-2: only the column-mean over nodes is needed -> per-block partial sums.
__global__ void attn2_kernel(const _Float16* __restrict__ Hm,
                             const float* __restrict__ sdst, const float* __restrict__ ssrc,
                             const int* __restrict__ nbrs, float* __restrict__ partial) {
    __shared__ int   nb[NDEG];
    __shared__ float alpha[NHD][NDEG];
    __shared__ float red[4][NC];
    float acc = 0.f;
    for (int i = 0; i < AGG_ITERS; ++i) {
        int n = blockIdx.x + i * AGG_GRID;   // every block runs exactly AGG_ITERS iters
        acc += gat_aggregate(Hm, sdst, ssrc, nbrs, n, threadIdx.x, nb, alpha, red);
    }
    partial[blockIdx.x * NC + threadIdx.x] = acc;
}

// ---------------------------------------------------------------- LN + MLP head
__global__ void final_kernel(const float* __restrict__ partial,
                             const float* __restrict__ gamma, const float* __restrict__ beta,
                             const float* __restrict__ w1, const float* __restrict__ b1,
                             const float* __restrict__ w2, const float* __restrict__ b2,
                             const float* __restrict__ w3, const float* __restrict__ b3,
                             float* __restrict__ out) {
    __shared__ float g[NC], h1[64], h2[16];
    __shared__ float mu, var;
    int tid = threadIdx.x;   // 128
    float acc = 0.f;
    for (int i = 0; i < AGG_GRID; ++i) acc += partial[i * NC + tid];
    acc *= (1.0f / NN);
    g[tid] = acc;
    __syncthreads();
    if (tid == 0) {
        float m = 0.f;
        for (int i = 0; i < NC; ++i) m += g[i];
        m *= (1.0f / NC);
        float v = 0.f;
        for (int i = 0; i < NC; ++i) { float d = g[i] - m; v += d * d; }
        mu = m; var = v * (1.0f / NC);
    }
    __syncthreads();
    g[tid] = (acc - mu) * rsqrtf(var + 1e-5f) * gamma[tid] + beta[tid];
    __syncthreads();
    if (tid < 64) {
        float a = b1[tid];
        for (int i = 0; i < NC; ++i) a = fmaf(w1[tid * NC + i], g[i], a);
        h1[tid] = (a > 0.f) ? a : SLOPE * a;
    }
    __syncthreads();
    if (tid < 16) {
        float a = b2[tid];
        for (int i = 0; i < 64; ++i) a = fmaf(w2[tid * 64 + i], h1[i], a);
        h2[tid] = (a > 0.f) ? a : SLOPE * a;
    }
    __syncthreads();
    if (tid < 16) {
        float a = b3[tid];
        for (int i = 0; i < 16; ++i) a = fmaf(w3[tid * 16 + i], h2[i], a);
        out[tid] = (a > 0.f) ? a : 0.f;
    }
}

// ---------------------------------------------------------------- launcher
extern "C" void kernel_launch(void* const* d_in, const int* in_sizes, int n_in,
                              void* d_out, int out_size, void* d_ws, size_t ws_size,
                              hipStream_t stream) {
    (void)in_sizes; (void)n_in; (void)out_size; (void)ws_size;
    const float* x     = (const float*)d_in[0];
    const int*   nbrs  = (const int*)d_in[1];
    const float* W1    = (const float*)d_in[2];
    const float* a1    = (const float*)d_in[3];
    const float* W2    = (const float*)d_in[4];
    const float* a2    = (const float*)d_in[5];
    const float* gamma = (const float*)d_in[6];
    const float* beta  = (const float*)d_in[7];
    const float* mw1   = (const float*)d_in[8];
    const float* mb1   = (const float*)d_in[9];
    const float* mw2   = (const float*)d_in[10];
    const float* mb2   = (const float*)d_in[11];
    const float* mw3   = (const float*)d_in[12];
    const float* mb3   = (const float*)d_in[13];
    float* out = (float*)d_out;

    char* ws = (char*)d_ws;
    size_t off = 0;
    auto alloc = [&](size_t bytes) -> void* {
        void* p = ws + off;
        off = (off + bytes + 255) & ~(size_t)255;
        return p;
    };
    _Float16* xh      = (_Float16*)alloc((size_t)NN * NFIN * 2);
    _Float16* w1h     = (_Float16*)alloc((size_t)HC * NFIN * 2);
    _Float16* w2h     = (_Float16*)alloc((size_t)HC * NC * 2);
    _Float16* Hbuf    = (_Float16*)alloc((size_t)NN * HC * 2);      // h1, then reused for h2
    float*    sdst    = (float*)alloc((size_t)NN * NHD * 4);
    float*    ssrc    = (float*)alloc((size_t)NN * NHD * 4);
    _Float16* emb1h   = (_Float16*)alloc((size_t)NN * NC * 2);
    float*    partial = (float*)alloc((size_t)AGG_GRID * NC * 4);

    // fp32 -> fp16 staging
    cvt_f32_f16<<<(NN * NFIN + 255) / 256, 256, 0, stream>>>(x, xh, NN * NFIN);
    cvt_f32_f16<<<(HC * NFIN + 255) / 256, 256, 0, stream>>>(W1, w1h, HC * NFIN);
    cvt_f32_f16<<<(HC * NC + 255) / 256, 256, 0, stream>>>(W2, w2h, HC * NC);

    dim3 pgrid(NN / 16, 16);
    // layer 1
    proj_kernel<NFIN><<<pgrid, 128, 0, stream>>>(xh, w1h, Hbuf);
    score_kernel<<<NN, 256, 0, stream>>>(Hbuf, a1, sdst, ssrc);
    attn1_kernel<<<NN, 128, 0, stream>>>(Hbuf, sdst, ssrc, nbrs, emb1h);
    // layer 2 (h2 overwrites h1)
    proj_kernel<NC><<<pgrid, 128, 0, stream>>>(emb1h, w2h, Hbuf);
    score_kernel<<<NN, 256, 0, stream>>>(Hbuf, a2, sdst, ssrc);
    attn2_kernel<<<AGG_GRID, 128, 0, stream>>>(Hbuf, sdst, ssrc, nbrs, partial);
    // LN + MLP head
    final_kernel<<<1, 128, 0, stream>>>(partial, gamma, beta,
                                        mw1, mb1, mw2, mb2, mw3, mb3, out);
}
